// SparseMoEBlock_vallina_9328668967106
// MI455X (gfx1250) — matrix operand order repaired and checked by reading, not compiled, not run on hardware
//
#include <hip/hip_runtime.h>
#include <math.h>

// ---------------- problem constants ----------------
constexpr int S    = 2048;   // B * SEQ
constexpr int D    = 768;
constexpr int E    = 8;
constexpr int H    = 3072;
constexpr int NTOT = E * S;  // 16384
constexpr int KSEL = 2 * S;  // CAPACITY * S = 4096

// ---------------- workspace layout (bytes) ----------------
constexpr size_t WS_SCORES = 0;                       // E*S f32   = 65536
constexpr size_t WS_GATE   = 65536;                   // E*S f32   = 65536
constexpr size_t WS_COUNTS = 131072;                  // E i32 (padded)
constexpr size_t WS_OFFS   = 131328;                  // E i32 (padded)
constexpr size_t WS_IDX    = 131584;                  // E*S i32   = 65536
constexpr size_t WS_GVAL   = 197120;                  // E*S f32   = 65536
constexpr size_t WS_HD     = 262656;                  // 4096*H bf16 = 25165824

// ---------------- WMMA types ----------------
typedef __attribute__((ext_vector_type(16))) __bf16 v16bf;
typedef __attribute__((ext_vector_type(8)))  __bf16 v8bf;
typedef __attribute__((ext_vector_type(8)))  float  v8f;

// GEMM tiling: block = 256 threads = 8 waves, tile 128x128, K-step 32.
// Wave w: rows (w&3)*32 (2 A-frags), cols (w>>2)*64 (4 B-frags) -> 8 WMMA/step.
constexpr int BM  = 128;
constexpr int BN  = 128;
constexpr int BK  = 32;
constexpr int LDT = BK + 8;   // padded LDS row stride (bf16) to spread banks

// A fragment (16x32 bf16, row-major in LDS, row stride LDT):
//  lane<16: M=lane, K=0..7 (elems 0-7) and K=16..23 (elems 8-15)
//  lane>=16: M=lane-16, K=8..15 and K=24..31
__device__ __forceinline__ v16bf load_a_frag(const __bf16* As, int rowBase, int lane) {
  int m  = lane & 15;
  int k0 = (lane >> 4) << 3;          // 0 or 8
  const __bf16* p = As + (rowBase + m) * LDT;
  v8bf lo = *(const v8bf*)(p + k0);
  v8bf hi = *(const v8bf*)(p + k0 + 16);
  return __builtin_shufflevector(lo, hi, 0,1,2,3,4,5,6,7,8,9,10,11,12,13,14,15);
}

// B fragment (32x16 bf16). Bt stored column-major: Bt[n*LDT + k].
//  lane<16: N=lane, K=0..7 and K=16..23 ; lane>=16: N=lane-16, K=8..15 and K=24..31
__device__ __forceinline__ v16bf load_b_frag(const __bf16* Bt, int colBase, int lane) {
  int n  = colBase + (lane & 15);
  int k0 = (lane >> 4) << 3;          // 0 or 8
  const __bf16* p = Bt + n * LDT + k0;
  v8bf lo = *(const v8bf*)(p);
  v8bf hi = *(const v8bf*)(p + 16);
  return __builtin_shufflevector(lo, hi, 0,1,2,3,4,5,6,7,8,9,10,11,12,13,14,15);
}

__device__ __forceinline__ v8bf cvt8(float4 a, float4 b) {
  v8bf r;
  r[0] = (__bf16)a.x; r[1] = (__bf16)a.y; r[2] = (__bf16)a.z; r[3] = (__bf16)a.w;
  r[4] = (__bf16)b.x; r[5] = (__bf16)b.y; r[6] = (__bf16)b.z; r[7] = (__bf16)b.w;
  return r;
}

// ---------------- kernel 0: zero output + counts ----------------
__global__ void moe_init_kernel(float* __restrict__ y, int* __restrict__ counts) {
  int i = blockIdx.x * blockDim.x + threadIdx.x;
  if (i < S * D) y[i] = 0.0f;
  if (i < E) counts[i] = 0;
}

// ---------------- kernel 1: gating logits + softmax ----------------
__global__ __launch_bounds__(256) void moe_gate_kernel(
    const float* __restrict__ x, const float* __restrict__ gw,
    float* __restrict__ scores) {
  int tok  = (blockIdx.x * blockDim.x + threadIdx.x) >> 5;
  int lane = threadIdx.x & 31;
  if (tok >= S) return;
  const float* xr = x + (size_t)tok * D;
  float acc[E];
#pragma unroll
  for (int e = 0; e < E; ++e) acc[e] = 0.0f;
  for (int d = lane; d < D; d += 32) {
    float xv = xr[d];
#pragma unroll
    for (int e = 0; e < E; ++e) acc[e] += xv * gw[e * D + d];
  }
#pragma unroll
  for (int off = 16; off >= 1; off >>= 1) {
#pragma unroll
    for (int e = 0; e < E; ++e) acc[e] += __shfl_xor(acc[e], off, 32);
  }
  if (lane == 0) {
    float m = acc[0];
#pragma unroll
    for (int e = 1; e < E; ++e) m = fmaxf(m, acc[e]);
    float ex[E], sum = 0.0f;
#pragma unroll
    for (int e = 0; e < E; ++e) { ex[e] = expf(acc[e] - m); sum += ex[e]; }
    float inv = 1.0f / sum;
#pragma unroll
    for (int e = 0; e < E; ++e) scores[e * S + tok] = ex[e] * inv;
  }
}

// ---------------- kernel 2: exact global top-KSEL by rank ----------------
__global__ __launch_bounds__(256) void moe_topk_kernel(
    const float* __restrict__ scores, float* __restrict__ gate) {
  __shared__ float sc[8192];
  int i = blockIdx.x * blockDim.x + threadIdx.x;  // 0..NTOT-1
  float v = scores[i];
  int rank = 0;
  for (int base = 0; base < NTOT; base += 8192) {
    __syncthreads();
    for (int j = threadIdx.x; j < 8192; j += blockDim.x) sc[j] = scores[base + j];
    __syncthreads();
#pragma unroll 4
    for (int j = 0; j < 8192; ++j) {
      float s = sc[j];
      int jg = base + j;
      rank += ((s > v) || (s == v && jg < i)) ? 1 : 0;
    }
  }
  gate[i] = (rank < KSEL) ? v : 0.0f;
}

// ---------------- kernel 3: per-expert compaction ----------------
__global__ void moe_route_kernel(const float* __restrict__ gate,
                                 int* __restrict__ counts,
                                 int* __restrict__ idx, float* __restrict__ gval) {
  int e = blockIdx.x;
  for (int s = threadIdx.x; s < S; s += blockDim.x) {
    float g = gate[e * S + s];
    if (g != 0.0f) {
      int p = atomicAdd(&counts[e], 1);
      idx[e * S + p]  = s;
      gval[e * S + p] = g;
    }
  }
}

// ---------------- kernel 4: tiny scan over E counts ----------------
__global__ void moe_scan_kernel(const int* __restrict__ counts, int* __restrict__ offs) {
  if (threadIdx.x == 0 && blockIdx.x == 0) {
    int acc = 0;
    for (int e = 0; e < E; ++e) { offs[e] = acc; acc += counts[e]; }
  }
}

// ---------------- kernel 5: GEMM1 (gathered X @ w1[e]) + bias + GELU -> Hd(bf16)
__global__ __launch_bounds__(256) void moe_gemm1_kernel(
    const float* __restrict__ x, const float* __restrict__ w1,
    const float* __restrict__ b1, const int* __restrict__ counts,
    const int* __restrict__ offs, const int* __restrict__ idx,
    __bf16* __restrict__ Hd) {
  int e    = blockIdx.z;
  int cnt  = counts[e];
  int row0 = blockIdx.y * BM;
  if (row0 >= cnt) return;
  int n0   = blockIdx.x * BN;
  int base = offs[e];
  const int*   idxl = idx + e * S;
  const float* w1e  = w1 + (size_t)e * D * H;

  __shared__ __align__(16) __bf16 As[BM * LDT];
  __shared__ __align__(16) __bf16 Bt[BN * LDT];

  int tid  = threadIdx.x;
  int lane = tid & 31;
  int w    = tid >> 5;
  int wr   = w & 3;    // row group (32 rows)
  int wc   = w >> 2;   // col half (64 cols)

  // A loader: thread -> (row, k-half of 16); clamped gather (garbage rows are
  // never stored: epilogue guards rg < cnt), so loads are branch-free.
  int ar   = tid >> 1;
  int akh  = (tid & 1) * 16;
  int argc = min(row0 + ar, cnt - 1);
  const float* xrow = x + (size_t)idxl[argc] * D;
  // B loader: thread -> 2 columns x 8 k-rows (transposed, contiguous b128 stores)
  int bn = (tid & 63) * 2;
  int bk = (tid >> 6) * 8;

  v8f c[2][4];
#pragma unroll
  for (int i = 0; i < 2; ++i)
#pragma unroll
    for (int j = 0; j < 4; ++j) c[i][j] = (v8f){0,0,0,0,0,0,0,0};

  for (int d0 = 0; d0 < D; d0 += BK) {
    __syncthreads();
    // ---- A tile: 4x float4 global load -> 2x b128 LDS store (packed cvt)
    const float* ap = xrow + d0 + akh;
    float4 a0 = *(const float4*)(ap);
    float4 a1 = *(const float4*)(ap + 4);
    float4 a2 = *(const float4*)(ap + 8);
    float4 a3 = *(const float4*)(ap + 12);
    *(v8bf*)(As + ar * LDT + akh)     = cvt8(a0, a1);
    *(v8bf*)(As + ar * LDT + akh + 8) = cvt8(a2, a3);
    // ---- B tile (transposed): 8x float2 rows -> 2 column vectors
    const float* bp = w1e + (size_t)(d0 + bk) * H + (n0 + bn);
    v8bf q0, q1;
#pragma unroll
    for (int u = 0; u < 8; ++u) {
      float2 tv = *(const float2*)(bp + (size_t)u * H);
      q0[u] = (__bf16)tv.x;
      q1[u] = (__bf16)tv.y;
    }
    *(v8bf*)(Bt + bn * LDT + bk)       = q0;
    *(v8bf*)(Bt + (bn + 1) * LDT + bk) = q1;
    if (d0 + BK < D) __builtin_prefetch(bp + (size_t)BK * H, 0, 0);
    __syncthreads();

    v16bf fa0 = load_a_frag(As, wr * 32,      lane);
    v16bf fa1 = load_a_frag(As, wr * 32 + 16, lane);
#pragma unroll
    for (int j = 0; j < 4; ++j) {
      v16bf fb = load_b_frag(Bt, wc * 64 + j * 16, lane);
      c[0][j] = __builtin_amdgcn_wmma_f32_16x16x32_bf16(false, fa0, false, fb,
                                                        (short)0, c[0][j], false, false);
      c[1][j] = __builtin_amdgcn_wmma_f32_16x16x32_bf16(false, fa1, false, fb,
                                                        (short)0, c[1][j], false, false);
    }
  }

  int mloc = (lane >> 4) * 8;
  int ncol = lane & 15;
#pragma unroll
  for (int j = 0; j < 4; ++j) {
    int col = n0 + wc * 64 + j * 16 + ncol;
    float bias = b1[e * H + col];
#pragma unroll
    for (int i = 0; i < 2; ++i) {
#pragma unroll
      for (int r = 0; r < 8; ++r) {
        int rg = row0 + wr * 32 + i * 16 + mloc + r;
        if (rg < cnt) {
          float hv = c[i][j][r] + bias;
          float t  = tanhf(0.7978845608028654f * (hv + 0.044715f * hv * hv * hv));
          Hd[(size_t)(base + rg) * H + col] = (__bf16)(0.5f * hv * (1.0f + t));
        }
      }
    }
  }
}

// ---------------- kernel 6: GEMM2 (Hd @ w2[e]) + bias, gated atomic scatter ----
__global__ __launch_bounds__(256) void moe_gemm2_kernel(
    const __bf16* __restrict__ Hd, const float* __restrict__ w2,
    const float* __restrict__ b2, const int* __restrict__ counts,
    const int* __restrict__ offs, const int* __restrict__ idx,
    const float* __restrict__ gval, float* __restrict__ y) {
  int e    = blockIdx.z;
  int cnt  = counts[e];
  int row0 = blockIdx.y * BM;
  if (row0 >= cnt) return;
  int n0   = blockIdx.x * BN;
  int base = offs[e];
  const int*   idxl  = idx  + e * S;
  const float* gvall = gval + e * S;
  const float* w2e   = w2 + (size_t)e * H * D;

  __shared__ __align__(16) __bf16 As[BM * LDT];
  __shared__ __align__(16) __bf16 Bt[BN * LDT];

  int tid  = threadIdx.x;
  int lane = tid & 31;
  int w    = tid >> 5;
  int wr   = w & 3;
  int wc   = w >> 2;

  // A loader: raw bf16 LDS<-global copy -> CDNA5 async-to-LDS path (ASYNCcnt).
  int ar   = tid >> 1;
  int akh  = (tid & 1) * 16;
  int argc = min(row0 + ar, cnt - 1);
  const __bf16* hrow = Hd + (size_t)(base + argc) * H;
  unsigned int lds_a = (unsigned int)(uintptr_t)(As + ar * LDT + akh);
  int bn = (tid & 63) * 2;
  int bk = (tid >> 6) * 8;

  v8f c[2][4];
#pragma unroll
  for (int i = 0; i < 2; ++i)
#pragma unroll
    for (int j = 0; j < 4; ++j) c[i][j] = (v8f){0,0,0,0,0,0,0,0};

  for (int k0 = 0; k0 < H; k0 += BK) {
    __syncthreads();
    // ---- A tile: 32 contiguous bytes per thread via async LDS DMA
    {
      const __bf16* gp = hrow + k0 + akh;
      asm volatile("global_load_async_to_lds_b128 %0, %1, off\n\t"
                   "global_load_async_to_lds_b128 %0, %1, off offset:16"
                   :: "v"(lds_a), "v"(gp) : "memory");
    }
    // ---- B tile (transposed): f32 -> bf16 packed
    const float* bp = w2e + (size_t)(k0 + bk) * D + (n0 + bn);
    v8bf q0, q1;
#pragma unroll
    for (int u = 0; u < 8; ++u) {
      float2 tv = *(const float2*)(bp + (size_t)u * D);
      q0[u] = (__bf16)tv.x;
      q1[u] = (__bf16)tv.y;
    }
    *(v8bf*)(Bt + bn * LDT + bk)       = q0;
    *(v8bf*)(Bt + (bn + 1) * LDT + bk) = q1;
    if (k0 + BK < H) __builtin_prefetch(bp + (size_t)BK * D, 0, 0);
    asm volatile("s_wait_asynccnt 0x0" ::: "memory");
    __syncthreads();

    v16bf fa0 = load_a_frag(As, wr * 32,      lane);
    v16bf fa1 = load_a_frag(As, wr * 32 + 16, lane);
#pragma unroll
    for (int j = 0; j < 4; ++j) {
      v16bf fb = load_b_frag(Bt, wc * 64 + j * 16, lane);
      c[0][j] = __builtin_amdgcn_wmma_f32_16x16x32_bf16(false, fa0, false, fb,
                                                        (short)0, c[0][j], false, false);
      c[1][j] = __builtin_amdgcn_wmma_f32_16x16x32_bf16(false, fa1, false, fb,
                                                        (short)0, c[1][j], false, false);
    }
  }

  int mloc = (lane >> 4) * 8;
  int ncol = lane & 15;
#pragma unroll
  for (int i = 0; i < 2; ++i) {
#pragma unroll
    for (int r = 0; r < 8; ++r) {
      int rg = row0 + wr * 32 + i * 16 + mloc + r;
      if (rg < cnt) {
        int   tok = idxl[rg];
        float gv  = gvall[rg];
        float* yr = y + (size_t)tok * D;
#pragma unroll
        for (int j = 0; j < 4; ++j) {
          int col = n0 + wc * 64 + j * 16 + ncol;
          atomicAdd(yr + col, gv * (c[i][j][r] + b2[e * D + col]));
        }
      }
    }
  }
}

// ---------------- launch ----------------
extern "C" void kernel_launch(void* const* d_in, const int* in_sizes, int n_in,
                              void* d_out, int out_size, void* d_ws, size_t ws_size,
                              hipStream_t stream) {
  const float* x  = (const float*)d_in[0];
  const float* gw = (const float*)d_in[1];
  const float* w1 = (const float*)d_in[2];
  const float* b1 = (const float*)d_in[3];
  const float* w2 = (const float*)d_in[4];
  const float* b2 = (const float*)d_in[5];
  float* y = (float*)d_out;

  char* ws = (char*)d_ws;
  float*  scores = (float*)(ws + WS_SCORES);
  float*  gate   = (float*)(ws + WS_GATE);
  int*    counts = (int*)(ws + WS_COUNTS);
  int*    offs   = (int*)(ws + WS_OFFS);
  int*    idx    = (int*)(ws + WS_IDX);
  float*  gval   = (float*)(ws + WS_GVAL);
  __bf16* Hd     = (__bf16*)(ws + WS_HD);

  moe_init_kernel<<<(S * D + 255) / 256, 256, 0, stream>>>(y, counts);
  moe_gate_kernel<<<S / 8, 256, 0, stream>>>(x, gw, scores);
  moe_topk_kernel<<<NTOT / 256, 256, 0, stream>>>(scores, gate);
  moe_route_kernel<<<E, 256, 0, stream>>>(gate, counts, idx, gval);
  moe_scan_kernel<<<1, 32, 0, stream>>>(counts, offs);

  dim3 g1(H / BN, (S + BM - 1) / BM, E);
  moe_gemm1_kernel<<<g1, 256, 0, stream>>>(x, w1, b1, counts, offs, idx, Hd);
  dim3 g2(D / BN, (S + BM - 1) / BM, E);
  moe_gemm2_kernel<<<g2, 256, 0, stream>>>(Hd, w2, b2, counts, offs, idx, gval, y);
}
